// DenseGCN_32160715112882
// MI455X (gfx1250) — compile-verified, matrix operand order checked by reading, not run on hardware
//
#include <hip/hip_runtime.h>
#include <hip/hip_bf16.h>

#define TPB 256
#define BN_SCALE 0.99999500003749977f
#define LEAKY 0.01f

typedef __attribute__((ext_vector_type(16))) __bf16       v16bf;
typedef __attribute__((ext_vector_type(8)))  float        v8f;
typedef __attribute__((ext_vector_type(8)))  unsigned int v8u;

__device__ __forceinline__ unsigned short f2bf(float f) {
  unsigned int u = __builtin_bit_cast(unsigned int, f);
  u = (u + 0x7FFFu + ((u >> 16) & 1u)) >> 16;
  return (unsigned short)u;
}
__device__ __forceinline__ float bf2f(unsigned short h) {
  unsigned int u = ((unsigned int)h) << 16;
  return __builtin_bit_cast(float, u);
}

// ---------------- LDS layout (bytes) ----------------
#define XS_OFF    0        // f32 [30][512]  x, later z (in place)
#define ABF_OFF   61440    // bf16 [32][512] A staging; later incep Bt0/Bt1 (2x f32[8][512])
#define HBF_OFF   94208    // bf16 [32][256] h; later incep Bt2 (f32[8][512])
#define OUTS_OFF  110592   // f32 [30][128]  SAGE2 out
#define ADJ_OFF   125952   // f32 [30][32]
#define DEG_OFF   129792   // f32 [32]
#define NORM_OFF  129920   // f32 [32]
#define ATTN_OFF  130048   // f32 [32]
#define GAP_OFF   130176   // f32 [16]
#define SMEM_BYTES 130240

struct Params {
  const float* x;
  const unsigned short* wt;   // bf16 transposed GEMM weights in ws
  const float* b1;            // gcn1_broot [256]
  const float* b2;            // gcn2_broot [128]
  const float* attnW;         // [128]
  const float* attnb;         // [1]
  const float* cw[10];        // incep0: bott,c0,c1,c2,res ; incep1: same
  const float* linW;          // [8][64]
  const float* linb;          // [64]
  float* out;                 // [B][3904]
};

// ws offsets (ushort elements): W1rel_T[256][512], W1root_T[256][512],
// W2rel_T[128][256], W2root_T[128][256]
#define W1A_OFF 0
#define W1B_OFF 131072
#define W2A_OFF 262144
#define W2B_OFF 294912
#define PREP_TOTAL 327680

__global__ __launch_bounds__(TPB) void prep_weights(const float* g1wr, const float* g1wo,
                                                    const float* g2wr, const float* g2wo,
                                                    unsigned short* wt) {
  int id = blockIdx.x * blockDim.x + threadIdx.x;
  if (id < 131072) {                       // W1rel^T [n=256][k=512]
    int n = id >> 9, k = id & 511;
    wt[W1A_OFF + id] = f2bf(g1wr[k * 256 + n]);
  } else if (id < 262144) {                // W1root^T
    int j = id - 131072; int n = j >> 9, k = j & 511;
    wt[W1B_OFF + j] = f2bf(g1wo[k * 256 + n]);
  } else if (id < 294912) {                // W2rel^T [n=128][k=256]
    int j = id - 262144; int n = j >> 8, k = j & 255;
    wt[W2A_OFF + j] = f2bf(g2wr[k * 128 + n]);
  } else if (id < PREP_TOTAL) {            // W2root^T
    int j = id - 294912; int n = j >> 8, k = j & 255;
    wt[W2B_OFF + j] = f2bf(g2wo[k * 128 + n]);
  }
}

// One accumulate pass over K:  acc[t] += A(16xK from LDS, tile mt) * B(Kx16, tiles ntBase+t)
// A in LDS row-major bf16 [32][K]; B in global row-major-transposed bf16 [N][K].
template<int NT, int K>
__device__ __forceinline__ void gemm_pass(const unsigned short* A_lds,
                                          const unsigned short* B_glb,
                                          int mt, int ntBase, int lane, v8f* acc) {
  const unsigned int* A32 = (const unsigned int*)A_lds;
  const unsigned int* B32 = (const unsigned int*)B_glb;
  const int nl = lane & 15;
  const int khalf = lane >> 4;
  const int Kd = K >> 1;                       // dwords per row
  const int arow = (mt * 16 + nl) * Kd;
#pragma unroll 4
  for (int ks = 0; ks < (K >> 5); ++ks) {
    const int i0 = ks * 16 + khalf * 4;        // dword offset of this lane's first K-group
    uint4 a0 = *(const uint4*)(A32 + arow + i0);
    uint4 a1 = *(const uint4*)(A32 + arow + i0 + 8);
    v8u au;
    au[0] = a0.x; au[1] = a0.y; au[2] = a0.z; au[3] = a0.w;
    au[4] = a1.x; au[5] = a1.y; au[6] = a1.z; au[7] = a1.w;
    v16bf av = __builtin_bit_cast(v16bf, au);
#pragma unroll
    for (int t = 0; t < NT; ++t) {
      const int brow = ((ntBase + t) * 16 + nl) * Kd;
      uint4 c0 = *(const uint4*)(B32 + brow + i0);
      uint4 c1 = *(const uint4*)(B32 + brow + i0 + 8);
      v8u bu;
      bu[0] = c0.x; bu[1] = c0.y; bu[2] = c0.z; bu[3] = c0.w;
      bu[4] = c1.x; bu[5] = c1.y; bu[6] = c1.z; bu[7] = c1.w;
      v16bf bv = __builtin_bit_cast(v16bf, bu);
      acc[t] = __builtin_amdgcn_wmma_f32_16x16x32_bf16(
          false, av, false, bv, (short)0, acc[t], false, false);
    }
  }
}

__device__ __forceinline__ void conv8_same(const float* in, float* out, const float* w,
                                           int K, int padl, const float* res, int tid) {
  for (int e = tid; e < 8 * 512; e += TPB) {
    const int c = e >> 9, l = e & 511;
    const float* wc = w + c * 8 * K;
    float s = 0.f;
#pragma unroll
    for (int i = 0; i < 8; ++i) {
      const float* inr = in + (i << 9);
      const float* wk = wc + i * K;
      for (int k = 0; k < K; ++k) {
        const int ll = l + k - padl;
        if ((unsigned)ll < 512u) s += inr[ll] * wk[k];
      }
    }
    out[e] = res ? (s + res[e]) : s;
  }
}

__global__ __launch_bounds__(TPB) void fused_gnn(Params p) {
  const int b = blockIdx.x;
  const int tid = threadIdx.x;
  const int lane = tid & 31;
  const int wv = tid >> 5;

  __shared__ __align__(16) unsigned char smem[SMEM_BYTES];
  float*          XS   = (float*)(smem + XS_OFF);
  unsigned short* ABF  = (unsigned short*)(smem + ABF_OFF);
  unsigned short* HBF  = (unsigned short*)(smem + HBF_OFF);
  float*          OUTS = (float*)(smem + OUTS_OFF);
  float*          ADJ  = (float*)(smem + ADJ_OFF);
  float*          DEG  = (float*)(smem + DEG_OFF);
  float*          NORM = (float*)(smem + NORM_OFF);
  float*          ATT  = (float*)(smem + ATTN_OFF);
  float*          GAP  = (float*)(smem + GAP_OFF);

  // ---- 1. load x[b] : [30][512] f32 ----
  {
    const float4* xin = (const float4*)(p.x + (size_t)b * 30 * 512);
    float4* xs4 = (float4*)XS;
    for (int i = tid; i < 30 * 512 / 4; i += TPB) xs4[i] = xin[i];
  }
  __syncthreads();

  // ---- 2. row norms (wave per row) ----
  for (int r = wv; r < 30; r += 8) {
    float s = 0.f;
    for (int d = lane; d < 512; d += 32) { float v = XS[r * 512 + d]; s += v * v; }
    for (int m = 16; m; m >>= 1) s += __shfl_xor(s, m);
    if (lane == 0) NORM[r] = fmaxf(sqrtf(s), 1e-12f);
  }
  __syncthreads();

  // ---- 3. cosine adjacency [30][30] ----
  for (int e = tid; e < 900; e += TPB) {
    const int n = e / 30, m = e % 30;
    const float* xr = XS + n * 512;
    const float* xc = XS + m * 512;
    float s = 0.f;
    for (int d = 0; d < 512; ++d) s += xr[d] * xc[d];
    ADJ[n * 32 + m] = s / (NORM[n] * NORM[m]);
  }
  __syncthreads();

  // ---- 4. top-10 per row + degree ----
  if (tid < 30) {
    float* row = ADJ + tid * 32;
    unsigned int sel = 0;
    for (int it = 0; it < 10; ++it) {
      float best = -3.4e38f; int bi = 0;
      for (int m = 0; m < 30; ++m) {
        if (sel & (1u << m)) continue;
        float v = row[m];
        if (v > best) { best = v; bi = m; }
      }
      sel |= 1u << bi;
    }
    float dsum = 0.f;
    for (int m = 0; m < 30; ++m) {
      if (!(sel & (1u << m))) row[m] = 0.f;
      dsum += row[m];
    }
    DEG[tid] = fmaxf(dsum, 1.0f);
  }
  __syncthreads();

  // ---- 5. aggX = (adj@x)/deg -> bf16 into ABF [32][512] (rows 30/31 zero) ----
  for (int e = tid; e < 32 * 512; e += TPB) {
    const int n = e >> 9, d = e & 511;
    float v = 0.f;
    if (n < 30) {
      const float* arow = ADJ + n * 32;
      float s = 0.f;
      for (int m = 0; m < 30; ++m) s += arow[m] * XS[m * 512 + d];
      v = s / DEG[n];
    }
    ABF[e] = f2bf(v);
  }
  __syncthreads();

  // ---- 6. SAGE1 GEMM: h = aggX@Wrel1 + x@Wroot1 + b1 ; leaky; -> HBF bf16 [32][256]
  const int mt1 = wv >> 2;            // waves 0-3: rows 0-15 ; 4-7: rows 16-31
  const int nb1 = (wv & 3) * 4;       // 4 N-tiles of 16 per wave (N=256)
  v8f acc1[4];
  {
    const v8f vz = {0.f, 0.f, 0.f, 0.f, 0.f, 0.f, 0.f, 0.f};
    acc1[0] = vz; acc1[1] = vz; acc1[2] = vz; acc1[3] = vz;
  }
  gemm_pass<4, 512>(ABF, p.wt + W1A_OFF, mt1, nb1, lane, acc1);
  __syncthreads();
  // rebuild A staging with raw x in bf16
  for (int e = tid; e < 32 * 512; e += TPB) {
    const int n = e >> 9, d = e & 511;
    ABF[e] = f2bf(n < 30 ? XS[n * 512 + d] : 0.f);
  }
  __syncthreads();
  gemm_pass<4, 512>(ABF, p.wt + W1B_OFF, mt1, nb1, lane, acc1);
  {
    const int khalf = lane >> 4, nl = lane & 15;
#pragma unroll
    for (int t = 0; t < 4; ++t) {
      const int n = (nb1 + t) * 16 + nl;
      const float bias = p.b1[n];
#pragma unroll
      for (int r = 0; r < 8; ++r) {
        const int m = mt1 * 16 + khalf * 8 + r;
        float v = acc1[t][r] + bias;
        v = v > 0.f ? v : LEAKY * v;
        HBF[m * 256 + n] = f2bf(v);
      }
    }
  }
  __syncthreads();

  // ---- 7. aggH = (adj@h)/deg -> bf16 into ABF [32][256] ----
  for (int e = tid; e < 32 * 256; e += TPB) {
    const int n = e >> 8, d = e & 255;
    float v = 0.f;
    if (n < 30) {
      const float* arow = ADJ + n * 32;
      float s = 0.f;
      for (int m = 0; m < 30; ++m) s += arow[m] * bf2f(HBF[m * 256 + d]);
      v = s / DEG[n];
    }
    ABF[e] = f2bf(v);
  }
  __syncthreads();

  // ---- 8. SAGE2 GEMM: out = aggH@Wrel2 + h@Wroot2 + b2 -> OUTS + d_out ----
  const int mt2 = wv >> 2;
  const int nb2 = (wv & 3) * 2;       // 2 N-tiles per wave (N=128)
  v8f acc2[2];
  {
    const v8f vz = {0.f, 0.f, 0.f, 0.f, 0.f, 0.f, 0.f, 0.f};
    acc2[0] = vz; acc2[1] = vz;
  }
  gemm_pass<2, 256>(ABF, p.wt + W2A_OFF, mt2, nb2, lane, acc2);
  gemm_pass<2, 256>(HBF, p.wt + W2B_OFF, mt2, nb2, lane, acc2);
  {
    const int khalf = lane >> 4, nl = lane & 15;
    float* outg = p.out + (size_t)b * 3904 + 64;
#pragma unroll
    for (int t = 0; t < 2; ++t) {
      const int n = (nb2 + t) * 16 + nl;
      const float bias = p.b2[n];
#pragma unroll
      for (int r = 0; r < 8; ++r) {
        const int m = mt2 * 16 + khalf * 8 + r;
        const float v = acc2[t][r] + bias;
        if (m < 30) { OUTS[m * 128 + n] = v; outg[m * 128 + n] = v; }
      }
    }
  }
  __syncthreads();

  // ---- 9. attention scalar per node; z = x * attn (in place over XS) ----
  if (tid < 30) {
    float s = p.attnb[0];
    for (int n = 0; n < 128; ++n) s += OUTS[tid * 128 + n] * p.attnW[n];
    ATT[tid] = s;
  }
  __syncthreads();
  for (int e = tid; e < 30 * 512; e += TPB) XS[e] *= ATT[e >> 9];
  __syncthreads();

  // ---- 10. Inception block 1 (Cin=30 from XS) ----
  float* Bt0 = (float*)(smem + ABF_OFF);            // [8][512]
  float* Bt1 = Bt0 + 8 * 512;                       // [8][512]
  float* Bt2 = (float*)(smem + HBF_OFF);            // [8][512] residual

  for (int e = tid; e < 8 * 512; e += TPB) {        // bottleneck + residual (K=1)
    const int c = e >> 9, l = e & 511;
    const float* wb = p.cw[0] + c * 30;
    const float* wr = p.cw[4] + c * 30;
    float sb = 0.f, sr = 0.f;
    for (int i = 0; i < 30; ++i) {
      const float z = XS[i * 512 + l];
      sb += z * wb[i]; sr += z * wr[i];
    }
    Bt0[e] = sb;
    const float rv = sr * BN_SCALE;
    Bt2[e] = rv > 0.f ? rv : 0.f;
  }
  __syncthreads();
  conv8_same(Bt0, Bt1, p.cw[1], 16, 7, nullptr, tid); __syncthreads();
  conv8_same(Bt1, Bt0, p.cw[2], 8, 3, nullptr, tid);  __syncthreads();
  conv8_same(Bt0, Bt1, p.cw[3], 4, 1, Bt2, tid);      __syncthreads();  // z1 in Bt1

  // ---- 11. Inception block 2 (Cin=8 from Bt1) ----
  for (int e = tid; e < 8 * 512; e += TPB) {
    const int c = e >> 9, l = e & 511;
    const float* wb = p.cw[5] + c * 8;
    const float* wr = p.cw[9] + c * 8;
    float sb = 0.f, sr = 0.f;
#pragma unroll
    for (int i = 0; i < 8; ++i) {
      const float z = Bt1[i * 512 + l];
      sb += z * wb[i]; sr += z * wr[i];
    }
    Bt0[e] = sb;
    const float rv = sr * BN_SCALE;
    Bt2[e] = rv > 0.f ? rv : 0.f;
  }
  __syncthreads();
  conv8_same(Bt0, Bt1, p.cw[6], 16, 7, nullptr, tid); __syncthreads();
  conv8_same(Bt1, Bt0, p.cw[7], 8, 3, nullptr, tid);  __syncthreads();
  conv8_same(Bt0, Bt1, p.cw[8], 4, 1, Bt2, tid);      __syncthreads();  // z2 in Bt1

  // ---- 12. GAP (wave per channel) + final linear ----
  if (wv < 8) {
    float s = 0.f;
    for (int l = lane; l < 512; l += 32) s += Bt1[wv * 512 + l];
    for (int m = 16; m; m >>= 1) s += __shfl_xor(s, m);
    if (lane == 0) GAP[wv] = s * (1.f / 512.f);
  }
  __syncthreads();
  if (tid < 64) {
    float s = p.linb[tid];
#pragma unroll
    for (int c = 0; c < 8; ++c) s += GAP[c] * p.linW[c * 64 + tid];
    p.out[(size_t)b * 3904 + tid] = s;
  }
}

extern "C" void kernel_launch(void* const* d_in, const int* in_sizes, int n_in,
                              void* d_out, int out_size, void* d_ws, size_t ws_size,
                              hipStream_t stream) {
  (void)out_size; (void)ws_size;
  // x has a unique flat size: 2048*30*512.  If it's input 0, the harness kept
  // dict-insertion order; otherwise assume jax tree order (sorted keys, x last).
  const int XFLAT = 2048 * 30 * 512;
  const bool xfirst = (n_in > 0 && in_sizes[0] == XFLAT);

  int i_x, i_g1wr, i_g1wo, i_g1b, i_g2wr, i_g2wo, i_g2b, i_aw, i_ab, i_lw, i_lb;
  int inc0, inc1;  // base indices of the 5 incep tensors (bott,c0,c1,c2,res)
  if (xfirst) {
    i_x = 0; i_g1wr = 1; i_g1wo = 2; i_g1b = 3; i_g2wr = 4; i_g2wo = 5; i_g2b = 6;
    i_aw = 7; i_ab = 8; inc0 = 9; inc1 = 14; i_lw = 19; i_lb = 20;
  } else {
    i_aw = 0; i_ab = 1; i_g1wr = 2; i_g1wo = 3; i_g1b = 4; i_g2wr = 5; i_g2wo = 6;
    i_g2b = 7; inc0 = 8; inc1 = 13; i_lw = 18; i_lb = 19; i_x = 20;
  }

  unsigned short* wt = (unsigned short*)d_ws;

  prep_weights<<<(PREP_TOTAL + TPB - 1) / TPB, TPB, 0, stream>>>(
      (const float*)d_in[i_g1wr], (const float*)d_in[i_g1wo],
      (const float*)d_in[i_g2wr], (const float*)d_in[i_g2wo], wt);

  Params p;
  p.x = (const float*)d_in[i_x];
  p.wt = wt;
  p.b1 = (const float*)d_in[i_g1b];
  p.b2 = (const float*)d_in[i_g2b];
  p.attnW = (const float*)d_in[i_aw];
  p.attnb = (const float*)d_in[i_ab];
  for (int j = 0; j < 5; ++j) p.cw[j] = (const float*)d_in[inc0 + j];
  for (int j = 0; j < 5; ++j) p.cw[5 + j] = (const float*)d_in[inc1 + j];
  p.linW = (const float*)d_in[i_lw];
  p.linb = (const float*)d_in[i_lb];
  p.out = (float*)d_out;

  fused_gnn<<<2048, TPB, 0, stream>>>(p);
}
